// IMUAdaptivePhysicTokenMixer_79216376807559
// MI455X (gfx1250) — matrix operand order repaired and checked by reading, not compiled, not run on hardware
//
#include <hip/hip_runtime.h>
#include <cstdint>
#include <cstddef>

// ---------------- CDNA5 vector types ----------------
typedef __attribute__((ext_vector_type(16))) __bf16 v16bf;
typedef __attribute__((ext_vector_type(8)))  float  v8f;

// ---------------- constants ----------------
#define TT 128
#define BB 128
#define CC 128
#define VV 5
#define NTOT (TT*BB)          // 16384
#define TWOC 256
#define V_TH 0.8f
#define INV_TAU 0.5f
#define EPS 1e-5f

// workspace offsets (bytes)
#define WS_W_OFF   0            // packed W: 2*8*8*512 ushorts = 131072 B
#define WS_MW_OFF  131072       // masked_w: 75 floats
#define WS_YGN_OFF 131584       // y_gn: 16384*128*5 floats = 41,943,040 B

// GEMM tile: per workgroup 16 n-values => 80 columns, all 128 output channels
#define XRAW_BYTES (16*1280*4)         // 81920 B raw f32 image of x block
#define XB_STRIDE  272                 // ushorts per B-column row (32B aligned rows)
#define XB_BYTES   (80*XB_STRIDE*2)    // 43520 B

__device__ __forceinline__ unsigned short f2bf(float x) {
  union { float f; unsigned u; } t; t.f = x;
  unsigned u = t.u;
  unsigned r = u + 0x7FFFu + ((u >> 16) & 1u);   // round-to-nearest-even
  return (unsigned short)(r >> 16);
}
__device__ __forceinline__ float bf2f(unsigned short h) {
  union { unsigned u; float f; } t; t.u = ((unsigned)h) << 16;
  return t.f;
}

// =====================================================================
// Kernel 1: pack W_fusion into WMMA A-fragment layout (bf16 hi/lo split)
//           + compute sigmoid-masked mixer weights
// A 16-bit 16x32 fragment layout (ISA 7.12.2):
//   lane l (m = l&15, hi = l>>4), element j (0..15):
//   K = (j<8) ? hi*8 + j : 16 + hi*8 + (j-8)
// =====================================================================
__global__ __launch_bounds__(256) void pack_kernel(
    const float* __restrict__ Wf,   // (128, 256)
    const float* __restrict__ Wm,   // (5,5,3)
    const float* __restrict__ adj,  // (5,5)
    unsigned short* __restrict__ wpack,
    float* __restrict__ mwout)
{
  int e    = blockIdx.x * 256 + threadIdx.x;   // 65536 elements
  int j    = e & 15;
  int lane = (e >> 4) & 31;
  int ks   = (e >> 9) & 7;
  int mt   = (e >> 12) & 7;
  int m    = mt * 16 + (lane & 15);
  int hi   = lane >> 4;
  int K    = (j < 8) ? (hi * 8 + j) : (16 + hi * 8 + (j - 8));
  int k    = ks * 32 + K;

  float val = Wf[m * TWOC + k];
  unsigned short h = f2bf(val);
  unsigned short l = f2bf(val - bf2f(h));
  wpack[((0 * 8 + mt) * 8 + ks) * 512 + lane * 16 + j] = h;   // hi fragments
  wpack[((8 + mt) * 8 + ks) * 512 + lane * 16 + j] = l;       // lo fragments

  if (blockIdx.x == 0 && threadIdx.x < 75) {
    int o = threadIdx.x / 15, r = threadIdx.x % 15, i = r / 3, kk = r % 3;
    float sym = 0.5f * (adj[o * 5 + i] + adj[i * 5 + o]);
    float sig = 1.f / (1.f + __expf(-sym));
    mwout[threadIdx.x] = Wm[o * 15 + i * 3 + kk] * sig;
  }
}

// =====================================================================
// Kernel 2: fusion GEMM (bf16 hi/lo WMMA) + GroupNorm1
// grid = 1024 workgroups x 256 threads (8 waves). WG covers n0..n0+15,
// all 128 output channels, 80 columns. K loop: 8 steps of 32.
// __launch_bounds__(256, 2): keep >=2 waves/SIMD so the 8-wave WG fits
// and VGPRs stay <=256 (no s_set_vgpr_msb churn).
// =====================================================================
__global__ __launch_bounds__(256, 2) void fusion_gemm_gn1(
    const float* __restrict__ x,            // (T,B,C,10)
    const unsigned short* __restrict__ wpack,
    const float* __restrict__ gn1w,
    const float* __restrict__ gn1b,
    float* __restrict__ ygn)                // (n,128,5)
{
  __shared__ __align__(16) char smem[XRAW_BYTES + XB_BYTES];
  float*          xraw = (float*)smem;                        // staged f32 block
  unsigned short* xb   = (unsigned short*)(smem + XRAW_BYTES);// bf16 B operand
  float*          ylds = (float*)smem;                        // reuses xraw region

  const int tid = threadIdx.x;
  const int n0  = blockIdx.x * 16;
  const float* src = x + (size_t)n0 * 1280;

  // ---- stage 80KB x block into LDS with gfx1250 async global->LDS DMA ----
#pragma unroll
  for (int r = 0; r < 20; ++r) {
    int byteOff = (r * 256 + tid) * 16;
    unsigned ldsOff = (unsigned)(uintptr_t)(void*)(smem + byteOff);
    const char* g = (const char*)src + byteOff;
    asm volatile("global_load_async_to_lds_b128 %0, %1, off"
                 :: "v"(ldsOff), "v"(g) : "memory");
  }
  asm volatile("s_wait_asynccnt 0x0" ::: "memory");
  __syncthreads();

  // ---- repack to bf16 B layout: xb[col][K], col=(ni,v), K=c (+128 -> gyr) ----
  for (int idx = tid; idx < 80 * 256; idx += 256) {
    int col = idx >> 8;          // 0..79
    int K   = idx & 255;         // 0..255
    int ni  = col / 5, v = col % 5;
    int c   = K & 127, half = K >> 7;
    float val = xraw[ni * 1280 + c * 10 + v + 5 * half];
    xb[col * XB_STRIDE + K] = f2bf(val);
  }
  __syncthreads();

  // ---- WMMA main loop ----
  const int w    = tid >> 5;       // wave -> M tile (o block of 16)
  const int lane = tid & 31;
  const int ll   = lane & 15;
  const int lh   = lane >> 4;

  v8f acc[5] = {};                 // 5 N-tiles of 16 cols each

#pragma unroll 2
  for (int ks = 0; ks < 8; ++ks) {
    v16bf aH = *(const v16bf*)(wpack + ((0 * 8 + w) * 8 + ks) * 512 + lane * 16);
    v16bf aL = *(const v16bf*)(wpack + ((8 + w) * 8 + ks) * 512 + lane * 16);
#pragma unroll
    for (int nt = 0; nt < 5; ++nt) {
      v16bf bf;
      __builtin_memcpy(&bf, xb + (nt * 16 + ll) * XB_STRIDE + ks * 32 + lh * 8, 16);
      __builtin_memcpy((char*)&bf + 16, xb + (nt * 16 + ll) * XB_STRIDE + ks * 32 + 16 + lh * 8, 16);
      acc[nt] = __builtin_amdgcn_wmma_f32_16x16x32_bf16(
          false, aH, false, bf, (short)0, acc[nt], false, false);
      acc[nt] = __builtin_amdgcn_wmma_f32_16x16x32_bf16(
          false, aL, false, bf, (short)0, acc[nt], false, false);
    }
  }

  // ---- stash y into LDS (ylds[col][o]) for GroupNorm ----
#pragma unroll
  for (int nt = 0; nt < 5; ++nt) {
#pragma unroll
    for (int r = 0; r < 8; ++r) {
      int o   = w * 16 + lh * 8 + r;     // C/D layout: M = r + 8*(lane>>4)
      int col = nt * 16 + ll;
      ylds[col * 128 + o] = acc[nt][r];
    }
  }
  __syncthreads();

  // ---- GroupNorm1: 16 groups of 16 threads, one n each ----
  const int g = tid >> 4;     // ni 0..15
  const int s = tid & 15;
  float sum = 0.f, sq = 0.f;
  for (int idx = s; idx < 640; idx += 16) {
    float vv = ylds[g * 640 + idx];      // idx = v*128 + o
    sum += vv; sq += vv * vv;
  }
#pragma unroll
  for (int msk = 8; msk >= 1; msk >>= 1) {
    sum += __shfl_xor(sum, msk, 32);
    sq  += __shfl_xor(sq,  msk, 32);
  }
  float mu = sum * (1.f / 640.f);
  float rs = rsqrtf(sq * (1.f / 640.f) - mu * mu + EPS);
  int n = n0 + g;
  for (int idx = s; idx < 640; idx += 16) {
    int v = idx >> 7;
    int o = idx & 127;
    float yv = (ylds[g * 640 + idx] - mu) * rs * gn1w[o] + gn1b[o];
    ygn[((size_t)n * 128 + o) * 5 + v] = yv;
  }
}

// =====================================================================
// Kernel 3: LIF1 (spikes -> packed bits) + masked conv over T + GN2 + LIF2
// one thread per (b,c) = 16384 threads
// =====================================================================
__device__ __forceinline__ float bit_at(unsigned b0, unsigned b1,
                                        unsigned b2, unsigned b3, int tt) {
  if (tt < 0 || tt >= 128) return 0.f;
  unsigned wsel = (unsigned)tt >> 5;
  unsigned word = (wsel == 0) ? b0 : (wsel == 1) ? b1 : (wsel == 2) ? b2 : b3;
  return (float)((word >> (tt & 31)) & 1u);
}

__global__ __launch_bounds__(256) void mixer_lif_kernel(
    const float* __restrict__ ygn,
    const float* __restrict__ mwg,    // (5,5,3) masked weights
    const float* __restrict__ gn2w,
    const float* __restrict__ gn2b,
    float* __restrict__ out)          // (T,B,C,10)
{
  int gt = blockIdx.x * 256 + threadIdx.x;   // 0..16383
  int b = gt >> 7, c = gt & 127;

  float mw[5][5][3];
#pragma unroll
  for (int i = 0; i < 75; ++i) ((float*)mw)[i] = mwg[i];

  // ---- LIF pass 1: produce packed spike bits per v ----
  unsigned bits[5][4] = {};
  float vm[5] = {0.f, 0.f, 0.f, 0.f, 0.f};
  for (int t = 0; t < 128; ++t) {
    const float* p = ygn + ((size_t)(t * 128 + b) * 128 + c) * 5;
#pragma unroll
    for (int v = 0; v < 5; ++v) {
      float xv = p[v];
      vm[v] += (xv - vm[v]) * INV_TAU;
      if (vm[v] >= V_TH) {
        bits[v][t >> 5] |= (1u << (t & 31));
        vm[v] = 0.f;
      }
    }
  }

  // ---- conv stats for GroupNorm2 ----
  float sum = 0.f, sq = 0.f;
  for (int t = 0; t < 128; ++t) {
#pragma unroll
    for (int o = 0; o < 5; ++o) {
      float a = 0.f;
#pragma unroll
      for (int i = 0; i < 5; ++i)
#pragma unroll
        for (int k = 0; k < 3; ++k)
          a += mw[o][i][k] * bit_at(bits[i][0], bits[i][1], bits[i][2], bits[i][3], t + k - 1);
      sum += a; sq += a * a;
    }
  }
  float mu = sum * (1.f / 640.f);
  float rs = rsqrtf(sq * (1.f / 640.f) - mu * mu + EPS);

  float w2[5], b2[5];
#pragma unroll
  for (int o = 0; o < 5; ++o) { w2[o] = gn2w[o]; b2[o] = gn2b[o]; }

  // ---- normalize + LIF pass 2 + duplicated output ----
  float vm2[5] = {0.f, 0.f, 0.f, 0.f, 0.f};
  for (int t = 0; t < 128; ++t) {
    float* po = out + ((size_t)(t * 128 + b) * 128 + c) * 10;
#pragma unroll
    for (int o = 0; o < 5; ++o) {
      float a = 0.f;
#pragma unroll
      for (int i = 0; i < 5; ++i)
#pragma unroll
        for (int k = 0; k < 3; ++k)
          a += mw[o][i][k] * bit_at(bits[i][0], bits[i][1], bits[i][2], bits[i][3], t + k - 1);
      float yv = (a - mu) * rs * w2[o] + b2[o];
      vm2[o] += (yv - vm2[o]) * INV_TAU;
      float sp = (vm2[o] >= V_TH) ? 1.f : 0.f;
      vm2[o] *= (1.f - sp);
      po[o]     = sp;
      po[o + 5] = sp;
    }
  }
}

// =====================================================================
extern "C" void kernel_launch(void* const* d_in, const int* in_sizes, int n_in,
                              void* d_out, int out_size, void* d_ws, size_t ws_size,
                              hipStream_t stream) {
  const float* x    = (const float*)d_in[0];  // x_spike (T,B,C,10)
  const float* Wf   = (const float*)d_in[1];  // W_fusion (C,2C,1)
  const float* gn1w = (const float*)d_in[2];
  const float* gn1b = (const float*)d_in[3];
  const float* Wm   = (const float*)d_in[4];  // W_mixer (5,5,3)
  const float* gn2w = (const float*)d_in[5];
  const float* gn2b = (const float*)d_in[6];
  const float* adj  = (const float*)d_in[7];  // (5,5)
  float* out = (float*)d_out;

  char* ws = (char*)d_ws;
  unsigned short* wpack = (unsigned short*)(ws + WS_W_OFF);
  float* mw  = (float*)(ws + WS_MW_OFF);
  float* ygn = (float*)(ws + WS_YGN_OFF);

  pack_kernel<<<256, 256, 0, stream>>>(Wf, Wm, adj, wpack, mw);
  fusion_gemm_gn1<<<NTOT / 16, 256, 0, stream>>>(x, wpack, gn1w, gn1b, ygn);
  mixer_lif_kernel<<<NTOT / 256, 256, 0, stream>>>(ygn, mw, gn2w, gn2b, out);
}
